// EmbeddingGNNAddGlobal_12326556139995
// MI455X (gfx1250) — matrix operand order repaired and checked by reading, not compiled, not run on hardware
//
#include <hip/hip_runtime.h>
#include <hip/hip_bf16.h>
#include <math.h>

// ---------------------------------------------------------------------------
// EmbeddingGNNAddGlobal forward for MI455X (gfx1250, wave32).
// Dense GEMMs: v_wmma_f32_16x16x32_f16. Block = 256 threads (8 waves),
// 64-row x 128-col C macro-tile: each wave owns one 16-col tile and 4 row
// tiles (4 accumulators). Software-pipelined, double-buffered LDS staging:
// per k-step, prefetch the next 64x32 A tile + 32x128 B tile group into
// registers (coalesced, clause-batched, clamp+select OOB handling - no exec
// divergence), run 4 WMMAs off the current LDS buffer, store prefetch to the
// alternate buffer, one workgroup barrier per step. 32-bit addressing
// throughout (all buffers < 2^31 bytes).
// Graph gather/scatter is streaming + f32 atomics (125MB feature matrix is
// resident in the 192MB L2).
// ---------------------------------------------------------------------------

typedef __attribute__((ext_vector_type(16))) _Float16 v16h;
typedef __attribute__((ext_vector_type(8)))  float    v8f;

#define N_NODES_C  100000
#define N_EDGES_C  400000
#define N_GRAPHS_C 4096
#define DFEAT_C    78

// ------------------------------- utility -----------------------------------

__global__ void fill_f32(float* __restrict__ p, float v, long long n) {
  long long i = (long long)blockIdx.x * blockDim.x + threadIdx.x;
  long long stride = (long long)gridDim.x * blockDim.x;
  for (; i < n; i += stride) p[i] = v;
}

__global__ void deg_accum(const int* __restrict__ dst, float* __restrict__ deg, int E) {
  int i = blockIdx.x * blockDim.x + threadIdx.x;
  if (i < E) atomicAdd(&deg[dst[i]], 1.0f);
}

__global__ void rsqrt_k(const float* __restrict__ deg, float* __restrict__ dinv, int n) {
  int i = blockIdx.x * blockDim.x + threadIdx.x;
  if (i < n) dinv[i] = rsqrtf(deg[i]);
}

// ------------------------------ WMMA GEMM ----------------------------------
// C[M,N] = A[M,K] @ B[K,N] (+bias) (+relu), row-major f32 in/out, f16 MACs.
// A fragment (16x32 f16): lane = m + 16*h, h = (k%16)>=8;
//   e = 2*((k%8)/2 + 4*(k>=16)) + (k%8)%2
// B fragment (32x16 f16): lane = 16*((k%16)>=8) + n%16; e = k%8 + 8*(k>=16).
// EXEC stays all-ones across WMMA (no early returns; OOB -> zero-pad in the
// pack via clamp+select, predicated epilogue stores).

__device__ __forceinline__ void load_tile_a(const float* __restrict__ A, int M, int K,
                                            int row0, int kb, int tid, bool fast,
                                            float va[8]) {
  if (fast) {
#pragma unroll
    for (int r = 0; r < 8; ++r) {
      int idx = tid + r * 256;                 // 0..2047
      int m = idx >> 5, k = idx & 31;
      va[r] = A[(row0 + m) * K + (kb + k)];
    }
  } else {
#pragma unroll
    for (int r = 0; r < 8; ++r) {
      int idx = tid + r * 256;
      int m = idx >> 5, k = idx & 31;
      bool ok = ((row0 + m) < M) & ((kb + k) < K);
      int off = ok ? ((row0 + m) * K + (kb + k)) : 0;   // clamped, always valid
      float v = A[off];
      va[r] = ok ? v : 0.0f;
    }
  }
}

__device__ __forceinline__ void load_tile_b(const float* __restrict__ B, int K, int N,
                                            int col0, int kb, int tid, bool fast,
                                            float vb[16]) {
  if (fast) {
#pragma unroll
    for (int i = 0; i < 16; ++i) {
      int idx = tid + i * 256;                 // 0..4095
      int k = idx >> 7, c = idx & 127;
      vb[i] = B[(kb + k) * N + (col0 + c)];
    }
  } else {
#pragma unroll
    for (int i = 0; i < 16; ++i) {
      int idx = tid + i * 256;
      int k = idx >> 7, c = idx & 127;
      bool ok = ((kb + k) < K) & ((col0 + c) < N);
      int off = ok ? ((kb + k) * N + (col0 + c)) : 0;
      float v = B[off];
      vb[i] = ok ? v : 0.0f;
    }
  }
}

__device__ __forceinline__ void store_tiles(_Float16* __restrict__ lA,
                                            _Float16* __restrict__ lB,
                                            int tid, const float va[8],
                                            const float vb[16]) {
#pragma unroll
  for (int r = 0; r < 8; ++r) {
    int idx = tid + r * 256;
    int m = idx >> 5, k = idx & 31;
    int kp = k & 15, h = kp >> 3, kpp = kp & 7;
    int e = (((kpp >> 1) + ((k >= 16) ? 4 : 0)) << 1) + (kpp & 1);
    lA[(m >> 4) * 512 + (((m & 15) + 16 * h) << 4) + e] = (_Float16)va[r];
  }
#pragma unroll
  for (int i = 0; i < 16; ++i) {
    int idx = tid + i * 256;
    int k = idx >> 7, c = idx & 127;
    int w = c >> 4, cl = c & 15;
    lB[((w * 32 + ((k >> 3) & 1) * 16 + cl) << 4) + ((k & 7) + ((k >> 4) << 3))] = (_Float16)vb[i];
  }
}

__global__ __launch_bounds__(256)
void wmma_gemm_f16(const float* __restrict__ A, const float* __restrict__ B,
                   const float* __restrict__ bias, float* __restrict__ C,
                   int M, int K, int N, int relu) {
  __shared__ __align__(64) _Float16 ldsA[2][4 * 512];   // [buf][rowtile][lane][e]
  __shared__ __align__(64) _Float16 ldsB[2][8 * 512];   // [buf][wave][lane][e]

  const int tid  = threadIdx.x;
  const int lane = tid & 31;
  const int wave = tid >> 5;
  const int half = lane >> 4;
  const int lm   = lane & 15;
  const int row0 = blockIdx.x * 64;
  const int col0 = blockIdx.y * 128;
  const int bcol = col0 + wave * 16 + lm;
  const bool fullM = (row0 + 64) <= M;
  const bool fullN = (col0 + 128) <= N;

  v8f acc0 = {}, acc1 = {}, acc2 = {}, acc3 = {};

  const int nk = (K + 31) >> 5;        // number of 32-wide k-steps
  float va[8], vb[16];

  // prologue: stage k-step 0 into buffer 0
  load_tile_a(A, M, K, row0, 0, tid, fullM && (32 <= K), va);
  load_tile_b(B, K, N, col0, 0, tid, fullN && (32 <= K), vb);
  store_tiles(ldsA[0], ldsB[0], tid, va, vb);

  int cur = 0;
  for (int s = 0; s < nk; ++s) {
    __syncthreads();   // stores to buf[cur] visible; prior reads of buf[cur^1] done

    // prefetch next k-step into registers (overlaps with the WMMAs below)
    const bool more = (s + 1) < nk;
    if (more) {
      int kb = (s + 1) << 5;
      load_tile_a(A, M, K, row0, kb, tid, fullM && (kb + 32 <= K), va);
      load_tile_b(B, K, N, col0, kb, tid, fullN && (kb + 32 <= K), vb);
    }

    // compute from buf[cur]
    const _Float16* lA = ldsA[cur];
    v16h b = *(const v16h*)(ldsB[cur] + ((wave * 32 + lane) << 4));
    {
      v16h a = *(const v16h*)(lA + 0 * 512 + (lane << 4));
      acc0 = __builtin_amdgcn_wmma_f32_16x16x32_f16(false, a, false, b,
                                                    (short)0, acc0, false, false);
    }
    {
      v16h a = *(const v16h*)(lA + 1 * 512 + (lane << 4));
      acc1 = __builtin_amdgcn_wmma_f32_16x16x32_f16(false, a, false, b,
                                                    (short)0, acc1, false, false);
    }
    {
      v16h a = *(const v16h*)(lA + 2 * 512 + (lane << 4));
      acc2 = __builtin_amdgcn_wmma_f32_16x16x32_f16(false, a, false, b,
                                                    (short)0, acc2, false, false);
    }
    {
      v16h a = *(const v16h*)(lA + 3 * 512 + (lane << 4));
      acc3 = __builtin_amdgcn_wmma_f32_16x16x32_f16(false, a, false, b,
                                                    (short)0, acc3, false, false);
    }

    // store prefetched tile into the alternate buffer
    if (more) store_tiles(ldsA[cur ^ 1], ldsB[cur ^ 1], tid, va, vb);
    cur ^= 1;
  }

  // ---- epilogue: D layout VGPR r -> M = r + 8*half, N = lm ----------------
  float bv = 0.0f;
  if (bias && bcol < N) bv = bias[bcol];
  v8f accs[4] = {acc0, acc1, acc2, acc3};
#pragma unroll
  for (int t = 0; t < 4; ++t) {
#pragma unroll
    for (int r = 0; r < 8; ++r) {
      int mrow = row0 + t * 16 + r + half * 8;
      if (mrow < M && bcol < N) {
        float v = accs[t][r] + bv;
        if (relu) v = fmaxf(v, 0.0f);
        C[mrow * N + bcol] = v;
      }
    }
  }
}

// --------------------------- graph operators -------------------------------

// agg[dst] += xw[src] * dinv[src]*dinv[dst]   (one block per edge)
__global__ void edge_scatter(const float* __restrict__ xw, const int* __restrict__ src,
                             const int* __restrict__ dst, const float* __restrict__ dinv,
                             float* __restrict__ agg, int F) {
  int e = blockIdx.x;
  int s = src[e], d = dst[e];
  float norm = dinv[s] * dinv[d];
  const float* xs = xw + (long long)s * F;
  float* ad = agg + (long long)d * F;
  for (int f = threadIdx.x; f < F; f += blockDim.x)
    atomicAdd(&ad[f], xs[f] * norm);
}

// x = relu(agg + xw*dinv^2 + b), in place over agg  (one block per node)
__global__ void gcn_combine(float* __restrict__ agg, const float* __restrict__ xw,
                            const float* __restrict__ dinv, const float* __restrict__ bias,
                            int F) {
  int node = blockIdx.x;
  float di = dinv[node];
  float d2 = di * di;
  long long base = (long long)node * F;
  for (int f = threadIdx.x; f < F; f += blockDim.x) {
    float v = agg[base + f] + xw[base + f] * d2 + bias[f];
    agg[base + f] = fmaxf(v, 0.0f);
  }
}

// sums[batch[n]] += x[n]; counts[batch[n]] += 1   (one block per node)
__global__ void pool_scatter(const float* __restrict__ x, const int* __restrict__ batch,
                             float* __restrict__ sums, float* __restrict__ counts, int F) {
  int node = blockIdx.x;
  int g = batch[node];
  if (threadIdx.x == 0) atomicAdd(&counts[g], 1.0f);
  long long bx = (long long)node * F;
  long long bs = (long long)g * F;
  for (int f = threadIdx.x; f < F; f += blockDim.x)
    atomicAdd(&sums[bs + f], x[bx + f]);
}

__global__ void pool_div(const float* __restrict__ sums, const float* __restrict__ counts,
                         float* __restrict__ pooled, int F, int G) {
  int i = blockIdx.x * blockDim.x + threadIdx.x;
  if (i < G * F) {
    int g = i / F;
    pooled[i] = sums[i] / fmaxf(counts[g], 1.0f);
  }
}

// gate = sigmoid([pooled, g] @ Wgate + b); fused = gate*g + (1-gate)*pooled
__global__ void gate_fuse(const float* __restrict__ pooled, const float* __restrict__ gv,
                          const float* __restrict__ Wgate, const float* __restrict__ bgate,
                          float* __restrict__ fused, int F, int G) {
  int i = blockIdx.x * blockDim.x + threadIdx.x;
  if (i >= G) return;
  const float* p  = pooled + (long long)i * F;
  const float* gg = gv     + (long long)i * F;
  float z = bgate[0];
  for (int j = 0; j < F; ++j) z += p[j]  * Wgate[j];
  for (int j = 0; j < F; ++j) z += gg[j] * Wgate[F + j];
  float gt = 1.0f / (1.0f + expf(-z));
  float* fo = fused + (long long)i * F;
  for (int j = 0; j < F; ++j) fo[j] = gt * gg[j] + (1.0f - gt) * p[j];
}

// ------------------------------- driver ------------------------------------

extern "C" void kernel_launch(void* const* d_in, const int* in_sizes, int n_in,
                              void* d_out, int out_size, void* d_ws, size_t ws_size,
                              hipStream_t stream) {
  (void)in_sizes; (void)n_in; (void)out_size; (void)ws_size;

  const float* mol_x      = (const float*)d_in[0];
  const int*   edge       = (const int*)d_in[1];
  const int*   src        = edge;
  const int*   dst        = edge + N_EDGES_C;
  const int*   batch      = (const int*)d_in[2];
  const float* global_emb = (const float*)d_in[3];
  const float* W1 = (const float*)d_in[4];  const float* b1 = (const float*)d_in[5];
  const float* W2 = (const float*)d_in[6];  const float* b2 = (const float*)d_in[7];
  const float* W3 = (const float*)d_in[8];  const float* b3 = (const float*)d_in[9];
  const float* Wfc1 = (const float*)d_in[10]; const float* bfc1 = (const float*)d_in[11];
  const float* Wfc2 = (const float*)d_in[12]; const float* bfc2 = (const float*)d_in[13];
  const float* Wproj = (const float*)d_in[14]; const float* bproj = (const float*)d_in[15];
  const float* Wgate = (const float*)d_in[16]; const float* bgate = (const float*)d_in[17];
  float* out = (float*)d_out;

  // workspace carve-up (256B aligned)
  char* ws = (char*)d_ws;
  size_t off = 0;
  auto carve = [&](size_t bytes) -> float* {
    float* p = (float*)(ws + off);
    off = (off + bytes + 255) & ~(size_t)255;
    return p;
  };
  const size_t nodeF = (size_t)N_NODES_C * 312 * sizeof(float);
  float* bufA   = carve(nodeF);                                   // xw
  float* bufB   = carve(nodeF);                                   // agg / activations
  float* deg    = carve((size_t)N_NODES_C * sizeof(float));
  float* dinv   = carve((size_t)N_NODES_C * sizeof(float));
  float* sums   = carve((size_t)N_GRAPHS_C * 312 * sizeof(float));
  float* counts = carve((size_t)N_GRAPHS_C * sizeof(float));
  float* pooled = carve((size_t)N_GRAPHS_C * 312 * sizeof(float));
  float* gvec   = carve((size_t)N_GRAPHS_C * 312 * sizeof(float));
  float* fused  = carve((size_t)N_GRAPHS_C * 312 * sizeof(float));
  float* hbuf   = carve((size_t)N_GRAPHS_C * 1024 * sizeof(float));

  // ---- degrees (with self loop) and normalization --------------------------
  fill_f32<<<1024, 256, 0, stream>>>(deg, 1.0f, (long long)N_NODES_C);
  deg_accum<<<(N_EDGES_C + 255) / 256, 256, 0, stream>>>(dst, deg, N_EDGES_C);
  rsqrt_k<<<(N_NODES_C + 255) / 256, 256, 0, stream>>>(deg, dinv, N_NODES_C);

  // ---- three GCN layers (bufB holds the layer output) ----------------------
  auto gcn_layer = [&](const float* xin, int Din, const float* W, const float* b, int Dout) {
    dim3 gg((N_NODES_C + 63) / 64, (Dout + 127) / 128);
    wmma_gemm_f16<<<gg, 256, 0, stream>>>(xin, W, nullptr, bufA,
                                          N_NODES_C, Din, Dout, 0);
    fill_f32<<<4096, 256, 0, stream>>>(bufB, 0.0f, (long long)N_NODES_C * Dout);
    edge_scatter<<<N_EDGES_C, 128, 0, stream>>>(bufA, src, dst, dinv, bufB, Dout);
    gcn_combine<<<N_NODES_C, 128, 0, stream>>>(bufB, bufA, dinv, b, Dout);
  };
  gcn_layer(mol_x, 78,  W1, b1, 78);
  gcn_layer(bufB,  78,  W2, b2, 156);   // GEMM reads bufB before it is zeroed
  gcn_layer(bufB,  156, W3, b3, 312);

  // ---- global mean pool ----------------------------------------------------
  fill_f32<<<1024, 256, 0, stream>>>(sums, 0.0f, (long long)N_GRAPHS_C * 312);
  fill_f32<<<64, 256, 0, stream>>>(counts, 0.0f, (long long)N_GRAPHS_C);
  pool_scatter<<<N_NODES_C, 128, 0, stream>>>(bufB, batch, sums, counts, 312);
  pool_div<<<(N_GRAPHS_C * 312 + 255) / 256, 256, 0, stream>>>(sums, counts, pooled,
                                                               312, N_GRAPHS_C);

  // ---- global-embedding projection + gated fusion --------------------------
  {
    dim3 gg(N_GRAPHS_C / 64, (312 + 127) / 128);
    wmma_gemm_f16<<<gg, 256, 0, stream>>>(global_emb, Wproj, bproj, gvec,
                                          N_GRAPHS_C, 128, 312, 0);
  }
  gate_fuse<<<(N_GRAPHS_C + 255) / 256, 256, 0, stream>>>(pooled, gvec, Wgate, bgate,
                                                          fused, 312, N_GRAPHS_C);

  // ---- MLP head ------------------------------------------------------------
  {
    dim3 g1(N_GRAPHS_C / 64, 1024 / 128);
    wmma_gemm_f16<<<g1, 256, 0, stream>>>(fused, Wfc1, bfc1, hbuf,
                                          N_GRAPHS_C, 312, 1024, 1);
    dim3 g2(N_GRAPHS_C / 64, 1);
    wmma_gemm_f16<<<g2, 256, 0, stream>>>(hbuf, Wfc2, bfc2, out,
                                          N_GRAPHS_C, 1024, 128, 0);
  }
}
